// get_model_26379689132426
// MI455X (gfx1250) — matrix operand order repaired and checked by reading, not compiled, not run on hardware
//
#include <hip/hip_runtime.h>
#include <cstdint>
#include <cstddef>

// ---------------------------------------------------------------------------
// CDNA5 (gfx1250) implementation of the PointNet++ / Mamba segmentation model.
// All GEMM-shaped work runs on v_wmma_f32_16x16x32_bf16 (f32 accumulate).
// ---------------------------------------------------------------------------

typedef __bf16 bf16;
typedef __attribute__((ext_vector_type(16))) __bf16 v16bf;
typedef __attribute__((ext_vector_type(8)))  __bf16 v8bf;
typedef __attribute__((ext_vector_type(8)))  float  v8f;

#define RES   25
#define RES3  (RES*RES*RES)
#define NB    2
#define NPTS  4096
#define DSTATE 16

__device__ __forceinline__ float siluf(float x)     { return x / (1.0f + __expf(-x)); }
__device__ __forceinline__ float softplusf(float x) { return (x > 20.0f) ? x : log1pf(__expf(x)); }

// ---------------------------------------------------------------------------
// WMMA bf16 GEMM:  Y[M,N](+ldy/col0) = act( X[M,K](+ldx/col0) @ W[K,N] + b )
// Block = 256 threads = 8 wave32s; block tile 128(M) x 64(N); K step 32.
// Each wave computes a 32x32 output (2 A frags x 2 B frags -> 4 WMMAs/K-step).
// LDS double-buffered, one barrier per K step. Staging is branchless in the
// fast path: out-of-range rows/cols are CLAMPED (their products only feed
// output rows/cols that are never stored), so no exec-mask juggling.
// act: 0=none 1=relu 2=softplus
// ---------------------------------------------------------------------------
__device__ __forceinline__ void gemm_stage(
    bf16* __restrict__ dA, bf16* __restrict__ dB,
    const float* __restrict__ X, int ldx, int xcol0,
    const float* __restrict__ W,
    int M, int N, int K, int m0, int n0, int k0, int tid)
{
  // ---- A tile 128x32: thread covers row r, cols c..c+15 (row clamped) ----
  {
    const int r = tid >> 1;
    const int c = (tid & 1) * 16;
    int gm = m0 + r; if (gm > M - 1) gm = M - 1;
    const float* s = X + (size_t)gm * ldx + xcol0 + k0 + c;
    v8bf av0 = {}, av1 = {};
    if (k0 + 32 <= K) {
#pragma unroll
      for (int j = 0; j < 8; ++j) av0[j] = (bf16)s[j];
#pragma unroll
      for (int j = 0; j < 8; ++j) av1[j] = (bf16)s[8 + j];
    } else {
#pragma unroll
      for (int j = 0; j < 8; ++j) {
        const int gk = k0 + c + j;
        av0[j] = (gk < K) ? (bf16)s[j] : (bf16)0.0f;
      }
#pragma unroll
      for (int j = 0; j < 8; ++j) {
        const int gk = k0 + c + 8 + j;
        av1[j] = (gk < K) ? (bf16)s[8 + j] : (bf16)0.0f;
      }
    }
    *(v8bf*)(dA + r*32 + c)     = av0;
    *(v8bf*)(dA + r*32 + c + 8) = av1;
  }
  // ---- B tile 64x32 (transposed to [n][k]): col clamped ----
  {
    const int n  = tid & 63;
    const int kc = (tid >> 6) * 8;
    int gn = n0 + n; if (gn > N - 1) gn = N - 1;
    const float* s = W + (size_t)(k0 + kc) * N + gn;
    v8bf bv = {};
    if (k0 + 32 <= K) {
#pragma unroll
      for (int j = 0; j < 8; ++j) bv[j] = (bf16)s[(size_t)j * N];
    } else {
#pragma unroll
      for (int j = 0; j < 8; ++j) {
        const int gk = k0 + kc + j;
        bv[j] = (gk < K) ? (bf16)s[(size_t)j * N] : (bf16)0.0f;
      }
    }
    *(v8bf*)(dB + n*32 + kc) = bv;
  }
}

__global__ __launch_bounds__(256) void k_gemm(
    const float* __restrict__ X, int ldx, int xcol0,
    const float* __restrict__ W, const float* __restrict__ bias,
    float* __restrict__ Y, int ldy, int ycol0,
    int M, int N, int K, int act)
{
  __shared__ alignas(64) bf16 sA[2][128*32];
  __shared__ alignas(64) bf16 sB[2][64*32];
  const int m0   = blockIdx.x * 128;
  const int n0   = blockIdx.y * 64;
  const int tid  = threadIdx.x;
  const int lane = tid & 31;
  const int wid  = tid >> 5;
  const int wm   = wid & 3;     // 0..3  -> 32-row M sub-block
  const int wn   = wid >> 2;    // 0..1  -> 32-col N sub-block
  const int half = lane >> 4;   // 0/1
  const int lr   = lane & 15;

  v8f acc[2][2] = {};           // [A tile 0/1][B tile 0/1]

  const int nk = (K + 31) >> 5;
  gemm_stage(sA[0], sB[0], X, ldx, xcol0, W, M, N, K, m0, n0, 0, tid);
  __syncthreads();

  for (int kt = 0; kt < nk; ++kt) {
    const int cur = kt & 1;
    if (kt + 1 < nk)
      gemm_stage(sA[cur ^ 1], sB[cur ^ 1], X, ldx, xcol0, W,
                 M, N, K, m0, n0, (kt + 1) * 32, tid);

    // A fragments (16-bit 16x32 layout): lanes 0-15 hold K{0..7,16..23},
    // lanes 16-31 hold K{8..15,24..31} of row M=lr.
    union { v16bf v; v8bf h[2]; } a0, a1;
    const bf16* ar0 = &sA[cur][(wm*32 +      lr)*32];
    const bf16* ar1 = &sA[cur][(wm*32 + 16 + lr)*32];
    a0.h[0] = *(const v8bf*)(ar0 + half*8);
    a0.h[1] = *(const v8bf*)(ar0 + 16 + half*8);
    a1.h[0] = *(const v8bf*)(ar1 + half*8);
    a1.h[1] = *(const v8bf*)(ar1 + 16 + half*8);

    // B fragments: lane n holds K{0..7,16..23} (half 0) / K{8..15,24..31} (half 1).
    union { v16bf v; v8bf h[2]; } bu0, bu1;
    const bf16* br0 = &sB[cur][(wn*32 +      lr)*32];
    const bf16* br1 = &sB[cur][(wn*32 + 16 + lr)*32];
    bu0.h[0] = *(const v8bf*)(br0 + half*8);
    bu0.h[1] = *(const v8bf*)(br0 + 16 + half*8);
    bu1.h[0] = *(const v8bf*)(br1 + half*8);
    bu1.h[1] = *(const v8bf*)(br1 + 16 + half*8);
    v16bf b0 = bu0.v;
    v16bf b1 = bu1.v;

    acc[0][0] = __builtin_amdgcn_wmma_f32_16x16x32_bf16(false, a0.v, false, b0,
                                                        (short)0, acc[0][0], false, false);
    acc[0][1] = __builtin_amdgcn_wmma_f32_16x16x32_bf16(false, a0.v, false, b1,
                                                        (short)0, acc[0][1], false, false);
    acc[1][0] = __builtin_amdgcn_wmma_f32_16x16x32_bf16(false, a1.v, false, b0,
                                                        (short)0, acc[1][0], false, false);
    acc[1][1] = __builtin_amdgcn_wmma_f32_16x16x32_bf16(false, a1.v, false, b1,
                                                        (short)0, acc[1][1], false, false);
    __syncthreads();
  }

  // C/D layout: lanes 0-15: VGPR r -> (M=r, N=lane); lanes 16-31: M=r+8.
  const int gmb = m0 + wm*32 + half*8;
  const int gnb = n0 + wn*32 + lr;
  if (m0 + 128 <= M && n0 + 64 <= N) {
#pragma unroll
    for (int i = 0; i < 2; ++i) {
#pragma unroll
      for (int r = 0; r < 8; ++r) {
        float* yp = Y + (size_t)(gmb + i*16 + r) * ldy + ycol0;
        float v0 = acc[i][0][r] + (bias ? bias[gnb]      : 0.0f);
        float v1 = acc[i][1][r] + (bias ? bias[gnb + 16] : 0.0f);
        if (act == 1)      { v0 = fmaxf(v0, 0.0f);  v1 = fmaxf(v1, 0.0f); }
        else if (act == 2) { v0 = softplusf(v0);    v1 = softplusf(v1);   }
        yp[gnb]      = v0;
        yp[gnb + 16] = v1;
      }
    }
  } else {
#pragma unroll
    for (int i = 0; i < 2; ++i) {
#pragma unroll
      for (int r = 0; r < 8; ++r) {
        const int gm = gmb + i*16 + r;
        if (gm >= M) continue;
#pragma unroll
        for (int j = 0; j < 2; ++j) {
          const int gn = gnb + j*16;
          if (gn >= N) continue;
          float v = acc[i][j][r] + (bias ? bias[gn] : 0.0f);
          if (act == 1)      v = fmaxf(v, 0.0f);
          else if (act == 2) v = softplusf(v);
          Y[(size_t)gm * ldy + ycol0 + gn] = v;
        }
      }
    }
  }
}

// ---------------------------------------------------------------------------
// Elementwise / gather / reduction kernels
// ---------------------------------------------------------------------------
__global__ void k_build_norm(const float* __restrict__ xyz, float* __restrict__ norm,
                             float* __restrict__ pts, int N) {
  size_t t = (size_t)blockIdx.x*blockDim.x + threadIdx.x;
  if (t >= (size_t)NB*N) return;
  int b = (int)(t / N), n = (int)(t % N);
  for (int c = 0; c < 8; ++c) {
    int row = (c < 4) ? c : c + 3;
    float v = xyz[((size_t)b*11 + row)*N + n];
    norm[t*8 + c] = v;
    if (c < 3) pts[t*3 + c] = v;
  }
}

__global__ void k_minmax(const float* __restrict__ pts, float* mn, float* mx, int N) {
  int b = threadIdx.x;
  if (b >= NB) return;
  float lo[3] = {1e30f,1e30f,1e30f}, hi[3] = {-1e30f,-1e30f,-1e30f};
  for (int n = 0; n < N; ++n)
    for (int d = 0; d < 3; ++d) {
      float v = pts[((size_t)b*N + n)*3 + d];
      lo[d] = fminf(lo[d], v); hi[d] = fmaxf(hi[d], v);
    }
  for (int d = 0; d < 3; ++d) { mn[b*3+d] = lo[d]; mx[b*3+d] = hi[d]; }
}

__global__ void k_voxel_accum(const float* __restrict__ pts, const float* __restrict__ norm,
                              const float* __restrict__ mn, const float* __restrict__ mx,
                              float* __restrict__ gfeat, float* __restrict__ gcnt,
                              int* __restrict__ flat, int N) {
  size_t t = (size_t)blockIdx.x*blockDim.x + threadIdx.x;
  if (t >= (size_t)NB*N) return;
  int b = (int)(t / N);
  int gi[3];
  for (int d = 0; d < 3; ++d) {
    float lo = mn[b*3+d], hi = mx[b*3+d];
    float g = (pts[t*3+d] - lo) / (hi - lo + 1e-8f) * (float)(RES - 1);
    int q = (int)rintf(g);
    gi[d] = q < 0 ? 0 : (q > RES-1 ? RES-1 : q);
  }
  int fl = (gi[0]*RES + gi[1])*RES + gi[2];
  flat[t] = fl;
  float* gp = gfeat + ((size_t)b*RES3 + fl)*8;
  for (int c = 0; c < 8; ++c) atomicAdd(gp + c, norm[t*8 + c]);
  atomicAdd(gcnt + (size_t)b*RES3 + fl, 1.0f);
}

__global__ void k_voxel_norm(float* __restrict__ gfeat, const float* __restrict__ gcnt) {
  size_t t = (size_t)blockIdx.x*blockDim.x + threadIdx.x;
  if (t >= (size_t)NB*RES3) return;
  float inv = 1.0f / fmaxf(gcnt[t], 1.0f);
  float* gp = gfeat + t*8;
  for (int c = 0; c < 8; ++c) gp[c] *= inv;
}

__global__ void k_conv3d(const float* __restrict__ grid, const float* __restrict__ W,
                         const float* __restrict__ bias, float* __restrict__ gf,
                         int kf, int cout, int co0) {
  size_t t = (size_t)blockIdx.x*blockDim.x + threadIdx.x;
  if (t >= (size_t)NB*RES3*cout) return;
  int co = (int)(t % cout);
  int v  = (int)((t / cout) % RES3);
  int b  = (int)(t / ((size_t)cout * RES3));
  int z = v % RES, y = (v / RES) % RES, x = v / (RES*RES);
  int pad = (kf - 1) / 2;
  float s = bias[co];
  for (int dx = 0; dx < kf; ++dx) {
    int xx = x + dx - pad; if ((unsigned)xx >= RES) continue;
    for (int dy = 0; dy < kf; ++dy) {
      int yy = y + dy - pad; if ((unsigned)yy >= RES) continue;
      for (int dz = 0; dz < kf; ++dz) {
        int zz = z + dz - pad; if ((unsigned)zz >= RES) continue;
        const float* gp = grid + ((size_t)b*RES3 + ((size_t)xx*RES + yy)*RES + zz)*8;
        const float* wp = W + (((size_t)dx*kf + dy)*kf + dz)*8*cout + co;
        for (int ci = 0; ci < 8; ++ci) s += gp[ci] * wp[(size_t)ci*cout];
      }
    }
  }
  gf[((size_t)b*RES3 + v)*64 + co0 + co] = fmaxf(s, 0.0f);
}

__global__ void k_gather_emb(const float* __restrict__ gf, const int* __restrict__ flat,
                             float* __restrict__ emb, int N) {
  size_t t = (size_t)blockIdx.x*blockDim.x + threadIdx.x;
  if (t >= (size_t)NB*N) return;
  int b = (int)(t / N);
  const float* sp = gf + ((size_t)b*RES3 + flat[t])*64;
  float* op = emb + t*64;
  for (int c = 0; c < 64; ++c) op[c] = sp[c];
}

__global__ void k_gather_rows(const float* __restrict__ src, int Ssrc, int D,
                              const int* __restrict__ idx, int idxStride, int Sdst,
                              float* __restrict__ out) {
  size_t t = (size_t)blockIdx.x*blockDim.x + threadIdx.x;
  if (t >= (size_t)NB*Sdst) return;
  int b = (int)(t / Sdst), s = (int)(t % Sdst);
  int g = idx[(size_t)b*idxStride + s];
  const float* sp = src + ((size_t)b*Ssrc + g)*D;
  float* op = out + t*(size_t)D;
  for (int c = 0; c < D; ++c) op[c] = sp[c];
}

__global__ void k_scatter_rows(const float* __restrict__ src, int S, int D,
                               const int* __restrict__ idx, int idxStride,
                               float* __restrict__ out) {
  size_t t = (size_t)blockIdx.x*blockDim.x + threadIdx.x;
  if (t >= (size_t)NB*S) return;
  int b = (int)(t / S), s = (int)(t % S);
  int g = idx[(size_t)b*idxStride + s];
  const float* sp = src + t*(size_t)D;
  float* op = out + ((size_t)b*S + g)*D;
  for (int c = 0; c < D; ++c) op[c] = sp[c];
}

__global__ void k_query_ball(const float* __restrict__ xyz, int Np,
                             const float* __restrict__ nxyz, int S,
                             float r2, int ns, int* __restrict__ gidx) {
  size_t t = (size_t)blockIdx.x*blockDim.x + threadIdx.x;
  if (t >= (size_t)NB*S) return;
  int b = (int)(t / S);
  float cx = nxyz[t*3+0], cy = nxyz[t*3+1], cz = nxyz[t*3+2];
  const float* xp = xyz + (size_t)b*Np*3;
  int* gp = gidx + t*(size_t)ns;
  int cnt = 0, first = 0; bool found = false;
  for (int j = 0; j < Np && cnt < ns; ++j) {
    float dx = xp[j*3+0]-cx, dy = xp[j*3+1]-cy, dz = xp[j*3+2]-cz;
    float d2 = dx*dx + dy*dy + dz*dz;
    if (d2 <= r2) { if (!found) { found = true; first = j; } gp[cnt++] = j; }
  }
  for (; cnt < ns; ++cnt) gp[cnt] = first;
}

__global__ void k_group(const float* __restrict__ feats, int C,
                        const float* __restrict__ xyz, int Np,
                        const float* __restrict__ nxyz,
                        const int* __restrict__ gidx, int S, int ns,
                        float* __restrict__ out) {
  size_t R = (size_t)NB*S*ns;
  size_t t = (size_t)blockIdx.x*blockDim.x + threadIdx.x;
  if (t >= R) return;
  int s = (int)((t / ns) % S);
  int b = (int)(t / ((size_t)ns*S));
  int g = gidx[t];
  const float* fp = feats + ((size_t)b*Np + g)*C;
  float* op = out + t*(size_t)(C + 3);
  for (int c = 0; c < C; ++c) op[c] = fp[c];
  for (int d = 0; d < 3; ++d)
    op[C+d] = xyz[((size_t)b*Np + g)*3 + d] - nxyz[((size_t)b*S + s)*3 + d];
}

__global__ void k_group_max(const float* __restrict__ in, int C, int ns, int S,
                            float* __restrict__ out, int ldo, int col0) {
  size_t t = (size_t)blockIdx.x*blockDim.x + threadIdx.x;
  if (t >= (size_t)NB*S*C) return;
  int c = (int)(t % C);
  int s = (int)((t / C) % S);
  int b = (int)(t / ((size_t)C*S));
  const float* ip = in + (((size_t)b*S + s)*ns)*C + c;
  float m = -1e30f;
  for (int j = 0; j < ns; ++j) m = fmaxf(m, ip[(size_t)j*C]);
  out[((size_t)b*S + s)*ldo + col0 + c] = m;
}

__global__ void k_colmax_scale(const float* __restrict__ in, int S, int C,
                               const float* __restrict__ scale,
                               float* __restrict__ out, int ldo, int col0) {
  size_t t = (size_t)blockIdx.x*blockDim.x + threadIdx.x;
  if (t >= (size_t)NB*C) return;
  int c = (int)(t % C), b = (int)(t / C);
  float m = -1e30f;
  for (int s = 0; s < S; ++s) m = fmaxf(m, in[((size_t)b*S + s)*C + c]);
  if (scale) m *= scale[0];
  out[(size_t)b*ldo + col0 + c] = m;
}

__global__ void k_layernorm(const float* __restrict__ x, int D,
                            const float* __restrict__ g, const float* __restrict__ bta,
                            float* __restrict__ out, int rows) {
  size_t r = (size_t)blockIdx.x*blockDim.x + threadIdx.x;
  if (r >= (size_t)rows) return;
  const float* xp = x + r*(size_t)D;
  float* op = out + r*(size_t)D;
  float m = 0.0f;
  for (int c = 0; c < D; ++c) m += xp[c];
  m /= (float)D;
  float v = 0.0f;
  for (int c = 0; c < D; ++c) { float d = xp[c] - m; v += d*d; }
  v /= (float)D;
  float inv = rsqrtf(v + 1e-5f);
  for (int c = 0; c < D; ++c) op[c] = (xp[c] - m)*inv*g[c] + bta[c];
}

__global__ void k_conv1d_silu(const float* __restrict__ xz, int ld, int L, int di,
                              const float* __restrict__ w, const float* __restrict__ cb,
                              float* __restrict__ xc) {
  size_t t = (size_t)blockIdx.x*blockDim.x + threadIdx.x;
  if (t >= (size_t)NB*L*di) return;
  int c = (int)(t % di);
  int l = (int)((t / di) % L);
  int b = (int)(t / ((size_t)di*L));
  float s = cb[c];
#pragma unroll
  for (int j = 0; j < 4; ++j) {
    int ll = l - 3 + j;
    if (ll >= 0) s += xz[((size_t)b*L + ll)*ld + c] * w[(size_t)j*di + c];
  }
  xc[((size_t)b*L + l)*di + c] = siluf(s);
}

__global__ void k_scan(const float* __restrict__ xc, const float* __restrict__ dt,
                       const float* __restrict__ dbc, int ldb, int dtr,
                       const float* __restrict__ A_log, const float* __restrict__ Dp,
                       float* __restrict__ y, int L, int di) {
  size_t t = (size_t)blockIdx.x*blockDim.x + threadIdx.x;
  if (t >= (size_t)NB*di) return;
  int c = (int)(t % di), b = (int)(t / di);
  float A[DSTATE], h[DSTATE];
#pragma unroll
  for (int s = 0; s < DSTATE; ++s) { A[s] = -__expf(A_log[(size_t)c*DSTATE + s]); h[s] = 0.0f; }
  float dp = Dp[c];
  for (int l = 0; l < L; ++l) {
    size_t row = (size_t)b*L + l;
    float dtt = dt[row*di + c];
    float xt  = xc[row*di + c];
    const float* Bt = dbc + row*ldb + dtr;
    const float* Ct = Bt + DSTATE;
    float acc = 0.0f;
#pragma unroll
    for (int s = 0; s < DSTATE; ++s) {
      h[s] = h[s]*__expf(dtt*A[s]) + dtt*xt*Bt[s];
      acc += h[s]*Ct[s];
    }
    y[row*di + c] = acc + xt*dp;
  }
}

__global__ void k_mul_silu(const float* __restrict__ y, const float* __restrict__ z,
                           int ldz, int zcol0, float* __restrict__ out,
                           int rows, int di) {
  size_t t = (size_t)blockIdx.x*blockDim.x + threadIdx.x;
  if (t >= (size_t)rows*di) return;
  size_t r = t / di; int c = (int)(t % di);
  out[t] = y[t] * siluf(z[r*ldz + zcol0 + c]);
}

__global__ void k_add(float* __restrict__ a, const float* __restrict__ b, size_t n) {
  size_t t = (size_t)blockIdx.x*blockDim.x + threadIdx.x;
  if (t < n) a[t] += b[t];
}

__global__ void k_copy_cols(const float* __restrict__ src, int lds_, int scol0,
                            float* __restrict__ dst, int ldd, int dcol0,
                            int rows, int cols) {
  size_t t = (size_t)blockIdx.x*blockDim.x + threadIdx.x;
  if (t >= (size_t)rows*cols) return;
  size_t r = t / cols; int c = (int)(t % cols);
  dst[r*ldd + dcol0 + c] = src[r*lds_ + scol0 + c];
}

__global__ void k_bcast(const float* __restrict__ glob, int Cg,
                        float* __restrict__ dst, int ldd, int dcol0, int rowsPerB) {
  size_t t = (size_t)blockIdx.x*blockDim.x + threadIdx.x;
  if (t >= (size_t)NB*rowsPerB) return;
  int b = (int)(t / rowsPerB);
  const float* gp = glob + (size_t)b*Cg;
  float* dp = dst + t*(size_t)ldd + dcol0;
  for (int c = 0; c < Cg; ++c) dp[c] = gp[c];
}

__global__ void k_three_interp(const float* __restrict__ xyz1, int N1,
                               const float* __restrict__ xyz2, int S2,
                               const float* __restrict__ f2, int C2,
                               float* __restrict__ dst, int ldd, int dcol0) {
  size_t t = (size_t)blockIdx.x*blockDim.x + threadIdx.x;
  if (t >= (size_t)NB*N1) return;
  int b = (int)(t / N1);
  float px = xyz1[t*3+0], py = xyz1[t*3+1], pz = xyz1[t*3+2];
  float bd[3] = {1e30f, 1e30f, 1e30f};
  int   bi[3] = {0, 0, 0};
  const float* x2 = xyz2 + (size_t)b*S2*3;
  for (int j = 0; j < S2; ++j) {
    float dx = px - x2[j*3+0], dy = py - x2[j*3+1], dz = pz - x2[j*3+2];
    float d2 = dx*dx + dy*dy + dz*dz;
    if (d2 < bd[0]) { bd[2]=bd[1]; bi[2]=bi[1]; bd[1]=bd[0]; bi[1]=bi[0]; bd[0]=d2; bi[0]=j; }
    else if (d2 < bd[1]) { bd[2]=bd[1]; bi[2]=bi[1]; bd[1]=d2; bi[1]=j; }
    else if (d2 < bd[2]) { bd[2]=d2; bi[2]=j; }
  }
  float w0 = 1.0f/(bd[0]+1e-8f), w1 = 1.0f/(bd[1]+1e-8f), w2 = 1.0f/(bd[2]+1e-8f);
  float ws = w0 + w1 + w2;
  w0 /= ws; w1 /= ws; w2 /= ws;
  const float* f0 = f2 + ((size_t)b*S2 + bi[0])*C2;
  const float* f1 = f2 + ((size_t)b*S2 + bi[1])*C2;
  const float* f3 = f2 + ((size_t)b*S2 + bi[2])*C2;
  float* dp = dst + t*(size_t)ldd + dcol0;
  for (int c = 0; c < C2; ++c) dp[c] = w0*f0[c] + w1*f1[c] + w2*f3[c];
}

__global__ void k_logsoftmax(const float* __restrict__ logits, int C,
                             float* __restrict__ out, int rows) {
  size_t r = (size_t)blockIdx.x*blockDim.x + threadIdx.x;
  if (r >= (size_t)rows) return;
  const float* lp = logits + r*(size_t)C;
  float m = -1e30f;
  for (int c = 0; c < C; ++c) m = fmaxf(m, lp[c]);
  float s = 0.0f;
  for (int c = 0; c < C; ++c) s += __expf(lp[c] - m);
  float ls = logf(s) + m;
  for (int c = 0; c < C; ++c) out[r*(size_t)C + c] = lp[c] - ls;
}

// ---------------------------------------------------------------------------
// Host-side helpers
// ---------------------------------------------------------------------------
struct Lin { const float* W; const float* b; };
struct MP  { const float *ln_g,*ln_b,*in_proj,*conv_w,*conv_b,*x_proj,*dt_w,*dt_b,*A_log,*Dp,*out_proj; };

struct Bump {
  char* base; size_t off;
  float* f(size_t n) { float* p = (float*)(base + off); off = (off + n*4 + 255) & ~(size_t)255; return p; }
  int*   i(size_t n) { int*   p = (int*)  (base + off); off = (off + n*4 + 255) & ~(size_t)255; return p; }
};

static inline dim3 g1d(size_t n) { return dim3((unsigned)((n + 255) / 256)); }

static void gemm_f(hipStream_t st, const float* X, int ldx, int xcol0,
                   const float* W, const float* bias,
                   float* Y, int ldy, int ycol0,
                   int M, int N, int K, int act) {
  dim3 grid((unsigned)((M + 127) / 128), (unsigned)((N + 63) / 64));
  k_gemm<<<grid, 256, 0, st>>>(X, ldx, xcol0, W, bias, Y, ldy, ycol0, M, N, K, act);
}

static float* mlp_chain(hipStream_t st, Bump& bp, const float* in, int rows, int cin,
                        const Lin* ls, const int* widths, int depth) {
  int wmax = 0;
  for (int d = 0; d < depth; ++d) if (widths[d] > wmax) wmax = widths[d];
  float* buf0 = bp.f((size_t)rows * wmax);
  float* buf1 = (depth > 1) ? bp.f((size_t)rows * wmax) : buf0;
  const float* x = in; int K = cin; float* y = buf0;
  for (int d = 0; d < depth; ++d) {
    gemm_f(st, x, K, 0, ls[d].W, ls[d].b, y, widths[d], 0, rows, widths[d], K, 1);
    x = y; K = widths[d];
    y = (y == buf0) ? buf1 : buf0;
  }
  return (float*)x;
}

static void sa_layer(hipStream_t st, Bump bp,
                     const float* xyz_in, int Np, const float* feats, int C,
                     const int* fps_idx, int fpsStride, int S,
                     const float* radii, const int* nsamp,
                     const Lin (*mlps)[3], const int (*widths)[3],
                     float* new_xyz, float* out, int outC) {
  k_gather_rows<<<g1d((size_t)NB*S), 256, 0, st>>>(xyz_in, Np, 3, fps_idx, fpsStride, S, new_xyz);
  int col0 = 0;
  for (int sc = 0; sc < 3; ++sc) {
    Bump sb = bp;                              // reuse stage scratch per scale
    int ns = nsamp[sc];
    size_t R = (size_t)NB * S * ns;
    int* gidx = sb.i(R);
    k_query_ball<<<g1d((size_t)NB*S), 256, 0, st>>>(xyz_in, Np, new_xyz, S,
                                                    radii[sc]*radii[sc], ns, gidx);
    float* grp = sb.f(R * (size_t)(C + 3));
    k_group<<<g1d(R), 256, 0, st>>>(feats, C, xyz_in, Np, new_xyz, gidx, S, ns, grp);
    float* last = mlp_chain(st, sb, grp, (int)R, C + 3, mlps[sc], widths[sc], 3);
    k_group_max<<<g1d((size_t)NB*S*widths[sc][2]), 256, 0, st>>>(
        last, widths[sc][2], ns, S, out, outC, col0);
    col0 += widths[sc][2];
  }
}

static void mamba_block(hipStream_t st, Bump bp, float* l, int S, int d,
                        const int* sidx, int sidxStride, const MP& p) {
  const int di = 2*d, dtr = d/16, L = S;
  const int rows = NB*L;
  const int nbc = dtr + 2*DSTATE;
  float* xg  = bp.f((size_t)rows*d);
  k_gather_rows<<<g1d((size_t)NB*L), 256, 0, st>>>(l, S, d, sidx, sidxStride, L, xg);
  float* xln = bp.f((size_t)rows*d);
  k_layernorm<<<g1d((size_t)rows), 256, 0, st>>>(xg, d, p.ln_g, p.ln_b, xln, rows);
  float* xz  = bp.f((size_t)rows*2*di);
  gemm_f(st, xln, d, 0, p.in_proj, nullptr, xz, 2*di, 0, rows, 2*di, d, 0);
  float* xc  = bp.f((size_t)rows*di);
  k_conv1d_silu<<<g1d((size_t)rows*di), 256, 0, st>>>(xz, 2*di, L, di, p.conv_w, p.conv_b, xc);
  float* dbc = bp.f((size_t)rows*nbc);
  gemm_f(st, xc, di, 0, p.x_proj, nullptr, dbc, nbc, 0, rows, nbc, di, 0);
  float* dt  = bp.f((size_t)rows*di);
  gemm_f(st, dbc, nbc, 0, p.dt_w, p.dt_b, dt, di, 0, rows, di, dtr, 2);
  float* y   = bp.f((size_t)rows*di);
  k_scan<<<g1d((size_t)NB*di), 256, 0, st>>>(xc, dt, dbc, nbc, dtr, p.A_log, p.Dp, y, L, di);
  float* ym  = bp.f((size_t)rows*di);
  k_mul_silu<<<g1d((size_t)rows*di), 256, 0, st>>>(y, xz, 2*di, di, ym, rows, di);
  float* mo  = bp.f((size_t)rows*d);
  gemm_f(st, ym, di, 0, p.out_proj, nullptr, mo, d, 0, rows, d, di, 0);
  k_add<<<g1d((size_t)rows*d), 256, 0, st>>>(xg, mo, (size_t)rows*d);
  k_scatter_rows<<<g1d((size_t)NB*L), 256, 0, st>>>(xg, L, d, sidx, sidxStride, l);
}

// ---------------------------------------------------------------------------
// Orchestration
// ---------------------------------------------------------------------------
extern "C" void kernel_launch(void* const* d_in, const int* in_sizes, int n_in,
                              void* d_out, int out_size, void* d_ws, size_t ws_size,
                              hipStream_t stream) {
  (void)in_sizes; (void)n_in; (void)out_size; (void)ws_size;
  int ti = 0;
  auto F = [&]() -> const float* { return (const float*)d_in[ti++]; };

  // inputs in setup_inputs() insertion order
  const float* in_xyz = F();
  const int* in_fps = (const int*)d_in[ti++];
  const int* in_ser = (const int*)d_in[ti++];
  const float *scW[3], *scB[3];
  for (int i = 0; i < 3; ++i) { scW[i] = F(); scB[i] = F(); }
  Lin sa1m[3][3], sa2m[3][3], sa3m[3][3], sa4m[3], fp4m[2], fp3m[2], fp2m[2], fp1m[2];
  for (int s = 0; s < 3; ++s) for (int d = 0; d < 3; ++d) sa1m[s][d] = {F(), F()};
  for (int s = 0; s < 3; ++s) for (int d = 0; d < 3; ++d) sa2m[s][d] = {F(), F()};
  for (int s = 0; s < 3; ++s) for (int d = 0; d < 3; ++d) sa3m[s][d] = {F(), F()};
  for (int d = 0; d < 3; ++d) sa4m[d] = {F(), F()};
  for (int d = 0; d < 2; ++d) fp4m[d] = {F(), F()};
  for (int d = 0; d < 2; ++d) fp3m[d] = {F(), F()};
  for (int d = 0; d < 2; ++d) fp2m[d] = {F(), F()};
  for (int d = 0; d < 2; ++d) fp1m[d] = {F(), F()};
  auto readMP = [&]() -> MP {
    MP m;
    m.ln_g = F(); m.ln_b = F(); m.in_proj = F(); m.conv_w = F(); m.conv_b = F();
    m.x_proj = F(); m.dt_w = F(); m.dt_b = F(); m.A_log = F(); m.Dp = F(); m.out_proj = F();
    return m;
  };
  MP m1 = readMP(), m2 = readMP(), m3 = readMP();
  const float* head_w1 = F(); const float* head_b1 = F();
  const float* head_w2 = F(); const float* head_b2 = F();
  const float* alpha = F(); const float* beta = F(); const float* gama = F();

  // persistent workspace buffers
  Bump ws{(char*)d_ws, 0};
  float* norm  = ws.f((size_t)NB*NPTS*8);
  float* pts   = ws.f((size_t)NB*NPTS*3);
  int*   flat  = ws.i((size_t)NB*NPTS);
  float* mnv   = ws.f(NB*3);
  float* mxv   = ws.f(NB*3);
  float* gfeat = ws.f((size_t)NB*RES3*8);
  float* gcnt  = ws.f((size_t)NB*RES3);
  float* gf    = ws.f((size_t)NB*RES3*64);
  float* emb   = ws.f((size_t)NB*NPTS*64);
  float* l1x   = ws.f((size_t)NB*1024*3);
  float* l1    = ws.f((size_t)NB*1024*320);
  float* l2x   = ws.f((size_t)NB*256*3);
  float* l2    = ws.f((size_t)NB*256*640);
  float* l3x   = ws.f((size_t)NB*64*3);
  float* l3    = ws.f((size_t)NB*64*1280);
  float* glob  = ws.f((size_t)NB*3264);
  float* l3f   = ws.f((size_t)NB*64*512);
  float* l2f   = ws.f((size_t)NB*256*256);
  float* l1f   = ws.f((size_t)NB*1024*128);
  float* l0    = ws.f((size_t)NB*NPTS*128);
  Bump arena = ws;                      // stage-local scratch starts here

  // ---- sparse CNN embedding ----
  k_build_norm<<<g1d((size_t)NB*NPTS), 256, 0, stream>>>(in_xyz, norm, pts, NPTS);
  k_minmax<<<1, NB, 0, stream>>>(pts, mnv, mxv, NPTS);
  (void)hipMemsetAsync(gfeat, 0, (size_t)NB*RES3*8*sizeof(float), stream);
  (void)hipMemsetAsync(gcnt,  0, (size_t)NB*RES3*sizeof(float), stream);
  k_voxel_accum<<<g1d((size_t)NB*NPTS), 256, 0, stream>>>(pts, norm, mnv, mxv,
                                                          gfeat, gcnt, flat, NPTS);
  k_voxel_norm<<<g1d((size_t)NB*RES3), 256, 0, stream>>>(gfeat, gcnt);
  {
    const int kfs[3] = {3,5,7}, cos[3] = {16,16,32}, co0s[3] = {0,16,32};
    for (int i = 0; i < 3; ++i)
      k_conv3d<<<g1d((size_t)NB*RES3*cos[i]), 256, 0, stream>>>(
          gfeat, scW[i], scB[i], gf, kfs[i], cos[i], co0s[i]);
  }
  k_gather_emb<<<g1d((size_t)NB*NPTS), 256, 0, stream>>>(gf, flat, emb, NPTS);

  // ---- SA1 + Mamba1 ----
  {
    const float radii[3] = {0.1f, 0.2f, 0.4f};
    const int   nsamp[3] = {32, 64, 128};
    const int   w[3][3]  = {{32,32,64},{64,64,128},{64,96,128}};
    sa_layer(stream, arena, pts, NPTS, emb, 64, in_fps + 0*1024, 3*1024, 1024,
             radii, nsamp, sa1m, w, l1x, l1, 320);
  }
  mamba_block(stream, arena, l1, 1024, 320, in_ser + 0*1024, 3*1024, m1);
  k_colmax_scale<<<g1d((size_t)NB*320), 256, 0, stream>>>(l1, 1024, 320, alpha, glob, 3264, 0);

  // ---- SA2 + Mamba2 ----
  {
    const float radii[3] = {0.2f, 0.4f, 0.8f};
    const int   nsamp[3] = {32, 64, 128};
    const int   w[3][3]  = {{64,64,128},{128,128,256},{128,128,256}};
    sa_layer(stream, arena, l1x, 1024, l1, 320, in_fps + 1*1024, 3*1024, 256,
             radii, nsamp, sa2m, w, l2x, l2, 640);
  }
  mamba_block(stream, arena, l2, 256, 640, in_ser + 1*1024, 3*1024, m2);
  k_colmax_scale<<<g1d((size_t)NB*640), 256, 0, stream>>>(l2, 256, 640, beta, glob, 3264, 320);

  // ---- SA3 + Mamba3 ----
  {
    const float radii[3] = {0.4f, 0.8f, 1.0f};
    const int   nsamp[3] = {16, 32, 64};
    const int   w[3][3]  = {{128,128,256},{256,256,512},{256,256,512}};
    sa_layer(stream, arena, l2x, 256, l2, 640, in_fps + 2*1024, 3*1024, 64,
             radii, nsamp, sa3m, w, l3x, l3, 1280);
  }
  mamba_block(stream, arena, l3, 64, 1280, in_ser + 2*1024, 3*1024, m3);
  k_colmax_scale<<<g1d((size_t)NB*1280), 256, 0, stream>>>(l3, 64, 1280, gama, glob, 3264, 960);

  // ---- SA4 global feature ----
  {
    Bump bp = arena;
    const int rows = NB*64;
    float* feat = bp.f((size_t)rows*1283);
    k_copy_cols<<<g1d((size_t)rows*3), 256, 0, stream>>>(l3x, 3, 0, feat, 1283, 0, rows, 3);
    k_copy_cols<<<g1d((size_t)rows*1280), 256, 0, stream>>>(l3, 1280, 0, feat, 1283, 3, rows, 1280);
    const int w4[3] = {256, 512, 1024};
    float* f4 = mlp_chain(stream, bp, feat, rows, 1283, sa4m, w4, 3);
    k_colmax_scale<<<g1d((size_t)NB*1024), 256, 0, stream>>>(f4, 64, 1024, nullptr, glob, 3264, 2240);
  }

  // ---- FP4 (S==1 broadcast) ----
  {
    Bump bp = arena;
    const int rows = NB*64;
    float* in4 = bp.f((size_t)rows*4544);
    k_copy_cols<<<g1d((size_t)rows*1280), 256, 0, stream>>>(l3, 1280, 0, in4, 4544, 0, rows, 1280);
    k_bcast<<<g1d((size_t)rows), 256, 0, stream>>>(glob, 3264, in4, 4544, 1280, 64);
    const int wf[2] = {1024, 512};
    float* o = mlp_chain(stream, bp, in4, rows, 4544, fp4m, wf, 2);
    k_copy_cols<<<g1d((size_t)rows*512), 256, 0, stream>>>(o, 512, 0, l3f, 512, 0, rows, 512);
  }

  // ---- FP3 ----
  {
    Bump bp = arena;
    const int rows = NB*256;
    float* in3 = bp.f((size_t)rows*1152);
    k_copy_cols<<<g1d((size_t)rows*640), 256, 0, stream>>>(l2, 640, 0, in3, 1152, 0, rows, 640);
    k_three_interp<<<g1d((size_t)NB*256), 256, 0, stream>>>(l2x, 256, l3x, 64, l3f, 512,
                                                            in3, 1152, 640);
    const int wf[2] = {512, 256};
    float* o = mlp_chain(stream, bp, in3, rows, 1152, fp3m, wf, 2);
    k_copy_cols<<<g1d((size_t)rows*256), 256, 0, stream>>>(o, 256, 0, l2f, 256, 0, rows, 256);
  }

  // ---- FP2 ----
  {
    Bump bp = arena;
    const int rows = NB*1024;
    float* in2 = bp.f((size_t)rows*576);
    k_copy_cols<<<g1d((size_t)rows*320), 256, 0, stream>>>(l1, 320, 0, in2, 576, 0, rows, 320);
    k_three_interp<<<g1d((size_t)NB*1024), 256, 0, stream>>>(l1x, 1024, l2x, 256, l2f, 256,
                                                             in2, 576, 320);
    const int wf[2] = {256, 128};
    float* o = mlp_chain(stream, bp, in2, rows, 576, fp2m, wf, 2);
    k_copy_cols<<<g1d((size_t)rows*128), 256, 0, stream>>>(o, 128, 0, l1f, 128, 0, rows, 128);
  }

  // ---- FP1 ----
  {
    Bump bp = arena;
    const int rows = NB*NPTS;
    float* in1 = bp.f((size_t)rows*195);
    k_copy_cols<<<g1d((size_t)rows*3), 256, 0, stream>>>(pts, 3, 0, in1, 195, 0, rows, 3);
    k_copy_cols<<<g1d((size_t)rows*64), 256, 0, stream>>>(emb, 64, 0, in1, 195, 3, rows, 64);
    k_three_interp<<<g1d((size_t)NB*NPTS), 256, 0, stream>>>(pts, NPTS, l1x, 1024, l1f, 128,
                                                             in1, 195, 67);
    const int wf[2] = {128, 128};
    float* o = mlp_chain(stream, bp, in1, rows, 195, fp1m, wf, 2);
    k_copy_cols<<<g1d((size_t)rows*128), 256, 0, stream>>>(o, 128, 0, l0, 128, 0, rows, 128);
  }

  // ---- head + log-softmax ----
  {
    Bump bp = arena;
    const int rows = NB*NPTS;
    float* h  = bp.f((size_t)rows*128);
    gemm_f(stream, l0, 128, 0, head_w1, head_b1, h, 128, 0, rows, 128, 128, 1);
    float* lg = bp.f((size_t)rows*13);
    gemm_f(stream, h, 128, 0, head_w2, head_b2, lg, 13, 0, rows, 13, 128, 0);
    k_logsoftmax<<<g1d((size_t)rows), 256, 0, stream>>>(lg, 13, (float*)d_out, rows);
  }
}
